// ScalarsMultiHeadSelfAttention_1975684956826
// MI455X (gfx1250) — compile-verified, matrix operand order checked by reading, not run on hardware
//
#include <hip/hip_runtime.h>
#include <hip/hip_bf16.h>
#include <stdint.h>

#define N_TOK 6144
#define E_DIM 64
#define H_NUM 8
#define D_DIM 512
#define NPAD  (N_TOK + 32)
#define QKV_COLS (3 * D_DIM)        // 1536
#define MT_TILES (N_TOK / 16)       // 384

typedef __attribute__((ext_vector_type(8)))  float    v8f;
typedef __attribute__((ext_vector_type(16))) __bf16   v16bf;

union Frag16 {
  v16bf bf;
  __bf16 h[16];
  uint4 q[2];
  unsigned short u[16];
};
union Acc8 {
  v8f   v;
  float f[8];
};
union Bf1 {
  __bf16 h;
  unsigned short u;
};
union Bf8 {
  __bf16 h[8];
  uint4 q;
};

// native f32 -> bf16 (round-to-nearest-even via hardware cvt)
__device__ __forceinline__ unsigned short f2bf(float x) {
  Bf1 p; p.h = (__bf16)x; return p.u;
}

// A-fragment (and symmetric B-fragment) load for 16x16x32 bf16 WMMA from a
// row-major bf16 (ushort) buffer. lane<16: elems 0..7 = [row, kb+0..7],
// elems 8..15 = [row, kb+16..23]; lanes 16..31 shift kb by +8.
__device__ __forceinline__ v16bf load_frag(const unsigned short* __restrict__ base,
                                           int row, int ld, int kb, int lane) {
  const int hi = (lane >> 4) & 1;
  const unsigned short* p = base + (size_t)row * (size_t)ld + kb + hi * 8;
  Frag16 f;
  f.q[0] = *(const uint4*)(p);
  f.q[1] = *(const uint4*)(p + 16);
  return f.bf;
}

__device__ __forceinline__ v8f wmma_bf16(v16bf a, v16bf b, v8f c) {
  return __builtin_amdgcn_wmma_f32_16x16x32_bf16(false, a, false, b, (short)0, c, false, false);
}

// ---------- segment bounds: batch sorted -> [lo,hi) per token --------------
__global__ void k_segments(const int* __restrict__ batch, int* __restrict__ lo,
                           int* __restrict__ hi) {
  int n = blockIdx.x * blockDim.x + threadIdx.x;
  if (n >= N_TOK) return;
  int v = batch[n];
  int a = 0, b = N_TOK;
  while (a < b) { int m = (a + b) >> 1; if (batch[m] < v) a = m + 1; else b = m; }
  lo[n] = a;
  a = 0; b = N_TOK;
  while (a < b) { int m = (a + b) >> 1; if (batch[m] <= v) a = m + 1; else b = m; }
  hi[n] = a;
}

// ---------- weight transposes to bf16 (column j contiguous along k) --------
__global__ void k_prep_wqkv(const float* __restrict__ W, unsigned short* __restrict__ WT) {
  int idx = blockIdx.x * blockDim.x + threadIdx.x;      // j*64 + k
  if (idx >= QKV_COLS * E_DIM) return;
  int j = idx >> 6, k = idx & 63;
  WT[idx] = f2bf(W[k * QKV_COLS + j]);
}
__global__ void k_prep_wout(const float* __restrict__ W, unsigned short* __restrict__ WT) {
  int idx = blockIdx.x * blockDim.x + threadIdx.x;      // j*512 + k
  if (idx >= E_DIM * D_DIM) return;
  int j = idx >> 9, k = idx & 511;
  WT[idx] = f2bf(W[k * E_DIM + j]);
}

// ---------- LN1 over E=64, one thread per row ------------------------------
__global__ void k_ln1(const float* __restrict__ x, const float* __restrict__ g,
                      const float* __restrict__ b, unsigned short* __restrict__ h1b) {
  int row = blockIdx.x * blockDim.x + threadIdx.x;
  if (row >= N_TOK) return;
  const float4* xr = (const float4*)(x + (size_t)row * 64);
  float vals[64];
  float s = 0.f;
#pragma unroll
  for (int i = 0; i < 16; ++i) {
    float4 t = xr[i];
    vals[4 * i + 0] = t.x; vals[4 * i + 1] = t.y;
    vals[4 * i + 2] = t.z; vals[4 * i + 3] = t.w;
    s += t.x + t.y + t.z + t.w;
  }
  float mean = s * (1.f / 64.f);
  float vs = 0.f;
#pragma unroll
  for (int i = 0; i < 64; ++i) { float d = vals[i] - mean; vs += d * d; }
  float inv = rsqrtf(vs * (1.f / 64.f) + 1e-5f);
  uint4* o = (uint4*)(h1b + (size_t)row * 64);
#pragma unroll
  for (int c = 0; c < 8; ++c) {
    Bf8 pk;
#pragma unroll
    for (int i = 0; i < 8; ++i) {
      int idx = c * 8 + i;
      pk.h[i] = (__bf16)((vals[idx] - mean) * inv * g[idx] + b[idx]);
    }
    o[c] = pk.q;
  }
}

// ---------- QKV GEMM: [N,64] @ [64,1536] + bias, WMMA bf16 -----------------
// Routes output into head-major Qb[h][n][64], Kb[h][n][64] (NPAD rows),
// and transposed Vt[h][64][NPAD].
__global__ void __launch_bounds__(128) k_qkv(const unsigned short* __restrict__ h1b,
    const unsigned short* __restrict__ WT, const float* __restrict__ bias,
    unsigned short* __restrict__ Qb, unsigned short* __restrict__ Kb,
    unsigned short* __restrict__ Vt) {
  int wid = blockIdx.x * (blockDim.x >> 5) + (threadIdx.x >> 5);
  const int numJT = QKV_COLS / 16;                      // 96
  int mt = wid / numJT, jt = wid - mt * numJT;
  if (mt >= MT_TILES) return;
  int lane = threadIdx.x & 31;
  int lo16 = lane & 15, hig = lane >> 4;

  int row = mt * 16 + lo16;
  v16bf a0 = load_frag(h1b, row, 64, 0, lane);
  v16bf a1 = load_frag(h1b, row, 64, 32, lane);
  int j = jt * 16 + lo16;
  v16bf b0 = load_frag(WT, j, 64, 0, lane);
  v16bf b1 = load_frag(WT, j, 64, 32, lane);

  v8f c = {0.f, 0.f, 0.f, 0.f, 0.f, 0.f, 0.f, 0.f};
  c = wmma_bf16(a0, b0, c);
  c = wmma_bf16(a1, b1, c);
  Acc8 acc; acc.v = c;

  float bj = bias[j];
  int sect = j >> 9;                 // 0 = K, 1 = Q, 2 = V (torch split order)
  int h = (j & 511) >> 6;
  int e = j & 63;
  int m0 = mt * 16 + 8 * hig;
  if (sect == 0) {
#pragma unroll
    for (int v = 0; v < 8; ++v)
      Kb[((size_t)h * NPAD + m0 + v) * 64 + e] = f2bf(acc.f[v] + bj);
  } else if (sect == 1) {
#pragma unroll
    for (int v = 0; v < 8; ++v)
      Qb[((size_t)h * N_TOK + m0 + v) * 64 + e] = f2bf(acc.f[v] + bj);
  } else {
    Bf8 pk;
#pragma unroll
    for (int v = 0; v < 8; ++v) pk.h[v] = (__bf16)(acc.f[v] + bj);
    *(uint4*)(Vt + ((size_t)h * 64 + e) * NPAD + m0) = pk.q;
  }
}

// ---------- flash attention per (head, 16-query tile), 1 wave/block --------
// Computes S^T = K Q^T via WMMA so softmax is (mostly) in-lane and the P^T
// fragment for O^T = V^T P^T needs no shuffles/LDS.
__global__ void __launch_bounds__(32) k_attn(const unsigned short* __restrict__ Qb,
    const unsigned short* __restrict__ Kb, const unsigned short* __restrict__ Vt,
    const int* __restrict__ lo, const int* __restrict__ hi,
    float* __restrict__ aout) {
  int h  = blockIdx.x / MT_TILES;
  int qt = blockIdx.x - h * MT_TILES;
  int lane = threadIdx.x;
  int lo16 = lane & 15, hig = lane >> 4;
  int q = qt * 16 + lo16;

  int sLo = lo[q], sHi = hi[q];
  int tlo = sLo, thi = sHi;
#pragma unroll
  for (int s = 1; s < 32; s <<= 1) {
    tlo = min(tlo, __shfl_xor(tlo, s, 32));
    thi = max(thi, __shfl_xor(thi, s, 32));
  }

  const unsigned short* Qh = Qb + (size_t)h * N_TOK * 64;
  const unsigned short* Kh = Kb + (size_t)h * NPAD * 64;
  const unsigned short* Vh = Vt + (size_t)h * 64 * NPAD;

  v16bf qb0 = load_frag(Qh, q, 64, 0, lane);   // B-frag: column q of Q^T
  v16bf qb1 = load_frag(Qh, q, 64, 32, lane);

  Acc8 o[4];
#pragma unroll
  for (int et = 0; et < 4; ++et)
#pragma unroll
    for (int v = 0; v < 8; ++v) o[et].f[v] = 0.f;
  float rmax = -3.0e38f, rsum = 0.f;
  const float scl = 0.04419417382415922f;      // 1/sqrt(512)

  for (int mb = (tlo & ~31); mb < thi; mb += 32) {
    Acc8 st[2];
#pragma unroll
    for (int t = 0; t < 2; ++t) {
      v16bf ka0 = load_frag(Kh, mb + t * 16 + lo16, 64, 0, lane);
      v16bf ka1 = load_frag(Kh, mb + t * 16 + lo16, 64, 32, lane);
      v8f c = {0.f, 0.f, 0.f, 0.f, 0.f, 0.f, 0.f, 0.f};
      c = wmma_bf16(ka0, qb0, c);
      c = wmma_bf16(ka1, qb1, c);
      st[t].v = c;
    }
    // scale + block-diagonal mask + chunk max (lane holds fixed q, 8 m's/tile)
    float cmax = -3.0e38f;
#pragma unroll
    for (int t = 0; t < 2; ++t)
#pragma unroll
      for (int v = 0; v < 8; ++v) {
        int m = mb + t * 16 + 8 * hig + v;
        float sv = (m >= sLo && m < sHi) ? st[t].f[v] * scl : -3.0e38f;
        st[t].f[v] = sv;
        cmax = fmaxf(cmax, sv);
      }
    cmax = fmaxf(cmax, __shfl_xor(cmax, 16, 32));
    float nmax = fmaxf(rmax, cmax);
    float corr = __expf(rmax - nmax);

    Frag16 pf;
    float csum = 0.f;
#pragma unroll
    for (int t = 0; t < 2; ++t)
#pragma unroll
      for (int v = 0; v < 8; ++v) {
        float sv = st[t].f[v];
        float p = (sv > -1.0e37f) ? __expf(sv - nmax) : 0.f;
        csum += p;
        pf.h[t * 8 + v] = (__bf16)p;           // lands exactly in B-frag order
      }
    csum += __shfl_xor(csum, 16, 32);
    rsum = rsum * corr + csum;
    rmax = nmax;

#pragma unroll
    for (int et = 0; et < 4; ++et) {
#pragma unroll
      for (int v = 0; v < 8; ++v) o[et].f[v] *= corr;
      v16bf va = load_frag(Vh, et * 16 + lo16, NPAD, mb, lane);
      o[et].v = wmma_bf16(va, pf.bf, o[et].v);
    }
  }

  float inv = 1.0f / rsum;                     // diagonal always valid
  float* dst = aout + (size_t)q * D_DIM + h * 64 + 8 * hig;
#pragma unroll
  for (int et = 0; et < 4; ++et) {
    float4 r0 = {o[et].f[0] * inv, o[et].f[1] * inv, o[et].f[2] * inv, o[et].f[3] * inv};
    float4 r1 = {o[et].f[4] * inv, o[et].f[5] * inv, o[et].f[6] * inv, o[et].f[7] * inv};
    *(float4*)(dst + et * 16)     = r0;
    *(float4*)(dst + et * 16 + 4) = r1;
  }
}

// ---------- LN2 over D=512, one wave per row -------------------------------
__global__ void __launch_bounds__(128) k_ln2(const float* __restrict__ aout,
    const float* __restrict__ g, const float* __restrict__ b,
    unsigned short* __restrict__ h2b) {
  int row = blockIdx.x * (blockDim.x >> 5) + (threadIdx.x >> 5);
  if (row >= N_TOK) return;
  int lane = threadIdx.x & 31;
  const float* r = aout + (size_t)row * 512 + lane * 16;
  float vals[16];
  float s = 0.f;
#pragma unroll
  for (int i = 0; i < 4; ++i) {
    float4 t = ((const float4*)r)[i];
    vals[4 * i + 0] = t.x; vals[4 * i + 1] = t.y;
    vals[4 * i + 2] = t.z; vals[4 * i + 3] = t.w;
    s += t.x + t.y + t.z + t.w;
  }
#pragma unroll
  for (int m = 1; m < 32; m <<= 1) s += __shfl_xor(s, m, 32);
  float mean = s * (1.f / 512.f);
  float vs = 0.f;
#pragma unroll
  for (int i = 0; i < 16; ++i) { float d = vals[i] - mean; vs += d * d; }
#pragma unroll
  for (int m = 1; m < 32; m <<= 1) vs += __shfl_xor(vs, m, 32);
  float inv = rsqrtf(vs * (1.f / 512.f) + 1e-5f);
  uint4* o = (uint4*)(h2b + (size_t)row * 512 + lane * 16);
#pragma unroll
  for (int half = 0; half < 2; ++half) {
    Bf8 pk;
#pragma unroll
    for (int i = 0; i < 8; ++i) {
      int c = lane * 16 + half * 8 + i;
      pk.h[i] = (__bf16)((vals[half * 8 + i] - mean) * inv * g[c] + b[c]);
    }
    o[half] = pk.q;
  }
}

// ---------- out projection: [N,512] @ [512,64] + bias, WMMA ----------------
__global__ void __launch_bounds__(128) k_out(const unsigned short* __restrict__ h2b,
    const unsigned short* __restrict__ WT, const float* __restrict__ bias,
    float* __restrict__ y) {
  int wid = blockIdx.x * (blockDim.x >> 5) + (threadIdx.x >> 5);
  int mt = wid >> 2, jt = wid & 3;
  if (mt >= MT_TILES) return;
  int lane = threadIdx.x & 31;
  int lo16 = lane & 15, hig = lane >> 4;
  int row = mt * 16 + lo16;
  int j = jt * 16 + lo16;
  v8f c = {0.f, 0.f, 0.f, 0.f, 0.f, 0.f, 0.f, 0.f};
#pragma unroll
  for (int kb = 0; kb < 512; kb += 32) {
    v16bf a = load_frag(h2b, row, 512, kb, lane);
    v16bf b = load_frag(WT, j, 512, kb, lane);
    c = wmma_bf16(a, b, c);
  }
  Acc8 acc; acc.v = c;
  float bj = bias[j];
  int m0 = mt * 16 + 8 * hig;
#pragma unroll
  for (int v = 0; v < 8; ++v)
    y[(size_t)(m0 + v) * 64 + j] = acc.f[v] + bj;
}

extern "C" void kernel_launch(void* const* d_in, const int* in_sizes, int n_in,
                              void* d_out, int out_size, void* d_ws, size_t ws_size,
                              hipStream_t stream) {
  (void)in_sizes; (void)n_in; (void)out_size; (void)ws_size;
  const float* x    = (const float*)d_in[0];
  const int* batch  = (const int*)d_in[1];
  const float* ln1g = (const float*)d_in[2];
  const float* ln1b = (const float*)d_in[3];
  const float* Wqkv = (const float*)d_in[4];
  const float* bqkv = (const float*)d_in[5];
  const float* ln2g = (const float*)d_in[6];
  const float* ln2b = (const float*)d_in[7];
  const float* Wout = (const float*)d_in[8];
  const float* bout = (const float*)d_in[9];
  float* y = (float*)d_out;

  char* ws = (char*)d_ws;
  size_t off = 0;
  auto take = [&](size_t bytes) -> char* {
    char* p = ws + off;
    off = (off + bytes + 255) & ~(size_t)255;
    return p;
  };
  int* lo = (int*)take((size_t)N_TOK * 4);
  int* hi = (int*)take((size_t)N_TOK * 4);
  unsigned short* h1b   = (unsigned short*)take((size_t)N_TOK * 64 * 2);
  unsigned short* WqkvT = (unsigned short*)take((size_t)QKV_COLS * 64 * 2);
  unsigned short* WoutT = (unsigned short*)take((size_t)64 * 512 * 2);
  unsigned short* Qb    = (unsigned short*)take((size_t)H_NUM * N_TOK * 64 * 2);
  unsigned short* Kb    = (unsigned short*)take((size_t)H_NUM * NPAD * 64 * 2);
  unsigned short* Vt    = (unsigned short*)take((size_t)H_NUM * 64 * NPAD * 2);
  float* aout           = (float*)take((size_t)N_TOK * D_DIM * 4);
  unsigned short* h2b   = (unsigned short*)take((size_t)N_TOK * D_DIM * 2);

  k_segments<<<(N_TOK + 255) / 256, 256, 0, stream>>>(batch, lo, hi);
  k_prep_wqkv<<<(QKV_COLS * 64 + 255) / 256, 256, 0, stream>>>(Wqkv, WqkvT);
  k_prep_wout<<<(64 * 512 + 255) / 256, 256, 0, stream>>>(Wout, WoutT);
  k_ln1<<<(N_TOK + 255) / 256, 256, 0, stream>>>(x, ln1g, ln1b, h1b);
  k_qkv<<<(MT_TILES * (QKV_COLS / 16)) / 4, 128, 0, stream>>>(h1b, WqkvT, bqkv, Qb, Kb, Vt);
  k_attn<<<H_NUM * MT_TILES, 32, 0, stream>>>(Qb, Kb, Vt, lo, hi, aout);
  k_ln2<<<N_TOK / 4, 128, 0, stream>>>(aout, ln2g, ln2b, h2b);
  k_out<<<MT_TILES, 128, 0, stream>>>(h2b, WoutT, bout, y);
}
